// GroupMoELayer_25486335935260
// MI455X (gfx1250) — compile-verified, matrix operand order checked by reading, not run on hardware
//
#include <hip/hip_runtime.h>
#include <hip/hip_bf16.h>
#include <cstdint>

typedef __bf16 bf16;
typedef bf16  v16bf __attribute__((ext_vector_type(16)));
typedef bf16  v8bf  __attribute__((ext_vector_type(8)));
typedef float v8f   __attribute__((ext_vector_type(8)));
typedef float v2f   __attribute__((ext_vector_type(2)));

#define BS    32768   // B*S tokens
#define HDIM  512
#define NEXP  32
#define DFF   1365
#define DFFP  1408    // 11*128, padded (zeros) for the down-GEMM K dim
#define KSEL  1024    // tokens per expert (bs*C_CAP/E)
#define KT    32      // bf16 GEMM k-tile

// ---------------------------------------------------------------- zero output
__global__ __launch_bounds__(256) void zero_f4(float4* __restrict__ p) {
  p[(size_t)blockIdx.x * 256 + threadIdx.x] = make_float4(0.f, 0.f, 0.f, 0.f);
}

// ------------------------------------------- router GEMM (f32 WMMA) + softmax
// M=BS, K=512, N=32.  Block: 256 thr = 8 waves, each wave owns 16 rows, N=32.
__global__ __launch_bounds__(256) void router_softmax(
    const float* __restrict__ x, const float* __restrict__ Wr,
    const float* __restrict__ br, float* __restrict__ gatesT) {
  const int tid = threadIdx.x, lane = tid & 31, w = tid >> 5;
  const int mb  = blockIdx.x * 128;
  const int row = mb + w * 16 + (lane & 15);
  __shared__ float Ws[64][32];          // staged W_router k-chunk (fp32)
  __shared__ float logits[8][16][32];
  v8f c0 = {}; v8f c1 = {};
  const int khalf = (lane >> 4) * 2;    // lanes 0-15: K0,K1 ; lanes 16-31: K2,K3
  for (int ko = 0; ko < HDIM; ko += 64) {
    for (int i = tid; i < 64 * 32; i += 256)
      Ws[i >> 5][i & 31] = Wr[(ko + (i >> 5)) * 32 + (i & 31)];
    __syncthreads();
#pragma unroll
    for (int kk = 0; kk < 64; kk += 4) {
      const int ks = kk + khalf;
      float2 av = *(const float2*)(x + (size_t)row * HDIM + ko + ks);
      v2f a;  a[0] = av.x;               a[1] = av.y;
      v2f b0; b0[0] = Ws[ks][lane & 15]; b0[1] = Ws[ks + 1][lane & 15];
      v2f b1; b1[0] = Ws[ks][16 + (lane & 15)]; b1[1] = Ws[ks + 1][16 + (lane & 15)];
      c0 = __builtin_amdgcn_wmma_f32_16x16x4_f32(false, a, false, b0, (short)0, c0, false, false);
      c1 = __builtin_amdgcn_wmma_f32_16x16x4_f32(false, a, false, b1, (short)0, c1, false, false);
    }
    __syncthreads();
  }
#pragma unroll
  for (int r = 0; r < 8; ++r) {         // C layout: lane=col, row = r + 8*(lane>=16)
    const int rr = (lane >> 4) * 8 + r;
    logits[w][rr][lane & 15]        = c0[r];
    logits[w][rr][16 + (lane & 15)] = c1[r];
  }
  __syncthreads();
  if (lane < 16) {                      // one lane per row: softmax over 32 experts
    const int token = mb + w * 16 + lane;
    float l[32], mx = -3.0e38f;
#pragma unroll
    for (int e = 0; e < 32; ++e) { l[e] = logits[w][lane][e] + br[e]; mx = fmaxf(mx, l[e]); }
    float s = 0.f;
#pragma unroll
    for (int e = 0; e < 32; ++e) { l[e] = __expf(l[e] - mx); s += l[e]; }
    const float inv = 1.f / s;
#pragma unroll
    for (int e = 0; e < 32; ++e) gatesT[(size_t)e * BS + token] = l[e] * inv;
  }
}

// ------------------------------------- per-expert top-k via 8-bit radix select
// Softmax gates are positive floats -> raw bits compare like uints (monotone).
__global__ __launch_bounds__(1024) void topk_select(
    const float* __restrict__ gatesT, int* __restrict__ idx_out,
    float* __restrict__ gval_out) {
  const int e = blockIdx.x, tid = threadIdx.x;
  const float* g = gatesT + (size_t)e * BS;
  __shared__ unsigned hist[256];
  __shared__ unsigned sGn[1024], sEn[1024];
  __shared__ unsigned sPrefix, sKt, sTotalG;
  unsigned pref = 0; int kt = KSEL;
  for (int pass = 0; pass < 4; ++pass) {
    const int shift = 24 - 8 * pass;
    if (tid < 256) hist[tid] = 0u;
    __syncthreads();
    const unsigned mask_hi = (pass == 0) ? 0u : (0xFFFFFFFFu << (shift + 8));
    for (int i = tid; i < BS; i += 1024) {
      const unsigned u = __float_as_uint(g[i]);
      if ((u & mask_hi) == pref) atomicAdd(&hist[(u >> shift) & 255], 1u);
    }
    __syncthreads();
    if (tid == 0) {
      int rk = kt; unsigned b = 0;
      for (int bin = 255; bin >= 0; --bin) {
        const int c = (int)hist[bin];
        if (rk <= c) { b = (unsigned)bin; break; }
        rk -= c;
      }
      sPrefix = pref | (b << shift); sKt = (unsigned)rk;
    }
    __syncthreads();
    pref = sPrefix; kt = (int)sKt;
    __syncthreads();
  }
  const unsigned T = pref;              // k-th largest value (bits)
  // deterministic compaction: contiguous 32-token chunk per thread + block scan
  const int base = tid * 32;
  int ng = 0, ne = 0;
  for (int i = 0; i < 32; ++i) {
    const unsigned u = __float_as_uint(g[base + i]);
    if (u > T) ++ng; else if (u == T) ++ne;
  }
  sGn[tid] = (unsigned)ng; sEn[tid] = (unsigned)ne;
  __syncthreads();
  if (tid == 0) {
    unsigned aG = 0, aE = 0;
    for (int i = 0; i < 1024; ++i) {
      const unsigned tg = sGn[i], te = sEn[i];
      sGn[i] = aG; sEn[i] = aE; aG += tg; aE += te;
    }
    sTotalG = aG;
  }
  __syncthreads();
  int gOff = (int)sGn[tid], eOff = (int)sEn[tid];
  const int totalG = (int)sTotalG, needE = KSEL - totalG;
  int*   oi = idx_out  + e * KSEL;
  float* og = gval_out + e * KSEL;
  for (int i = 0; i < 32; ++i) {
    const int t = base + i;
    const unsigned u = __float_as_uint(g[t]);
    if (u > T)      { oi[gOff] = t; og[gOff] = g[t]; ++gOff; }
    else if (u == T){ if (eOff < needE) { oi[totalG + eOff] = t; og[totalG + eOff] = g[t]; } ++eOff; }
  }
}

// ----------------------------- bf16 fragment loader from K-contiguous LDS tile
__device__ inline v16bf load_frag16(const bf16* __restrict__ tile, int row16, int lane) {
  const bf16* p = tile + (row16 + (lane & 15)) * KT;
  const int h8 = (lane >> 4) * 8;       // lanes 0-15: K 0-7/16-23, 16-31: K 8-15/24-31
  v8bf lo = *(const v8bf*)(p + h8);
  v8bf hi = *(const v8bf*)(p + 16 + h8);
  v16bf o;
#pragma unroll
  for (int i = 0; i < 8; ++i) { o[i] = lo[i]; o[i + 8] = hi[i]; }
  return o;
}

// ------------------ expert up-projection (gather, bf16 WMMA) + bias + SiLU
// Per block: 128(M) x 128(N) tile; 8 waves as 2(M) x 4(N); 8 WMMA accs/wave.
__global__ __launch_bounds__(256) void up_silu(
    const float* __restrict__ x, const float* __restrict__ W_up,
    const float* __restrict__ b_up, const int* __restrict__ idx,
    bf16* __restrict__ act) {
  const int e = blockIdx.z, grp = e >> 2;
  const int nb = blockIdx.x * 128, mb = blockIdx.y * 128;
  const int tid = threadIdx.x, lane = tid & 31, w = tid >> 5;
  const int wm = w & 1, wn = w >> 1;
  __shared__ __align__(16) bf16 As[128 * KT];
  __shared__ __align__(16) bf16 Bs[128 * KT];
  __shared__ int tok[128];
  if (tid < 128) tok[tid] = idx[e * KSEL + mb + tid];
  __syncthreads();
  v8f vzero = {};
  v8f acc[4][2];
#pragma unroll
  for (int f = 0; f < 4; ++f)
#pragma unroll
    for (int j = 0; j < 2; ++j) acc[f][j] = vzero;
  const float* Wg = W_up + (size_t)grp * HDIM * DFF;
  const int r = tid >> 1, kofs = (tid & 1) * 16;
  for (int k0 = 0; k0 < HDIM; k0 += KT) {
    { // A: gather token rows, fp32 -> bf16
      const float* src = x + (size_t)tok[r] * HDIM + k0 + kofs;
      float4 f0 = ((const float4*)src)[0], f1 = ((const float4*)src)[1];
      float4 f2 = ((const float4*)src)[2], f3 = ((const float4*)src)[3];
      v8bf o0, o1;
      o0[0]=(bf16)f0.x; o0[1]=(bf16)f0.y; o0[2]=(bf16)f0.z; o0[3]=(bf16)f0.w;
      o0[4]=(bf16)f1.x; o0[5]=(bf16)f1.y; o0[6]=(bf16)f1.z; o0[7]=(bf16)f1.w;
      o1[0]=(bf16)f2.x; o1[1]=(bf16)f2.y; o1[2]=(bf16)f2.z; o1[3]=(bf16)f2.w;
      o1[4]=(bf16)f3.x; o1[5]=(bf16)f3.y; o1[6]=(bf16)f3.z; o1[7]=(bf16)f3.w;
      *(v8bf*)&As[r * KT + kofs]     = o0;
      *(v8bf*)&As[r * KT + kofs + 8] = o1;
    }
    { // B: coalesced global read along N, transposed into LDS (N-major), bf16
      const int n = nb + (tid & 127), kh = tid >> 7;
#pragma unroll
      for (int it = 0; it < 16; ++it) {
        const int kk = it * 2 + kh;
        const float v = (n < DFF) ? Wg[(size_t)(k0 + kk) * DFF + n] : 0.f;
        Bs[(tid & 127) * KT + kk] = (bf16)v;
      }
      // prefetch next k-tile of the weight stream (global_prefetch_b8)
      if (k0 + KT < HDIM && n < DFF)
        __builtin_prefetch(Wg + (size_t)(k0 + KT + kh) * DFF + n, 0, 1);
    }
    __syncthreads();
    v16bf af[4], bf2[2];
#pragma unroll
    for (int f = 0; f < 4; ++f) af[f] = load_frag16(As, wm * 64 + f * 16, lane);
#pragma unroll
    for (int j = 0; j < 2; ++j) bf2[j] = load_frag16(Bs, wn * 32 + j * 16, lane);
#pragma unroll
    for (int f = 0; f < 4; ++f)
#pragma unroll
      for (int j = 0; j < 2; ++j)
        acc[f][j] = __builtin_amdgcn_wmma_f32_16x16x32_bf16(
            false, af[f], false, bf2[j], (short)0, acc[f][j], false, false);
    __syncthreads();
  }
#pragma unroll
  for (int j = 0; j < 2; ++j) {
    const int n = nb + wn * 32 + j * 16 + (lane & 15);
    const float bias = (n < DFF) ? b_up[grp * DFF + n] : 0.f;
#pragma unroll
    for (int f = 0; f < 4; ++f) {
      const int m0 = mb + wm * 64 + f * 16 + (lane >> 4) * 8;
#pragma unroll
      for (int rr = 0; rr < 8; ++rr) {
        const float v = acc[f][j][rr] + bias;
        const float s = (n < DFF) ? (v / (1.f + __expf(-v))) : 0.f;  // SiLU; pad cols = 0
        act[(size_t)(e * KSEL + m0 + rr) * DFFP + n] = (bf16)s;
      }
    }
  }
}

// ------ expert down-projection (bf16 WMMA) + bias + gated scatter-add combine
// A tile (bf16 activations) staged with CDNA5 async global->LDS copies
// (ASYNCcnt path): no VGPR round-trip, completion via s_wait_asynccnt.
__global__ __launch_bounds__(256) void down_combine(
    const bf16* __restrict__ act, const float* __restrict__ W_down,
    const float* __restrict__ b_down, const int* __restrict__ idx,
    const float* __restrict__ gval, float* __restrict__ y) {
  const int e = blockIdx.z;
  const int nb = blockIdx.x * 128, mb = blockIdx.y * 128;
  const int tid = threadIdx.x, lane = tid & 31, w = tid >> 5;
  const int wm = w & 1, wn = w >> 1;
  __shared__ __align__(16) bf16 As[128 * KT];
  __shared__ __align__(16) bf16 Bs[128 * KT];
  __shared__ int tok[128];
  __shared__ float gv[128];
  if (tid < 128) { tok[tid] = idx[e * KSEL + mb + tid]; gv[tid] = gval[e * KSEL + mb + tid]; }
  __syncthreads();
  v8f vzero = {};
  v8f acc[4][2];
#pragma unroll
  for (int f = 0; f < 4; ++f)
#pragma unroll
    for (int j = 0; j < 2; ++j) acc[f][j] = vzero;
  const int r = tid >> 1, kofs = (tid & 1) * 16;
  const bf16* Arow = act + (size_t)(e * KSEL + mb + r) * DFFP;
  // LDS byte offset of this thread's A-tile slot (low 32 bits of flat shared addr)
  const unsigned ldsA = (unsigned)(size_t)&As[r * KT + kofs];
  for (int k0 = 0; k0 < DFFP; k0 += KT) {   // K padded with zeros -> exact
    { // A: async DMA global -> LDS, 2 x 16B per thread (offset applies to both sides)
      const void* srcA = (const void*)(Arow + k0 + kofs);
      asm volatile("global_load_async_to_lds_b128 %0, %1, off"
                   :: "v"(ldsA), "v"(srcA) : "memory");
      asm volatile("global_load_async_to_lds_b128 %0, %1, off offset:16"
                   :: "v"(ldsA), "v"(srcA) : "memory");
    }
    { // B: coalesced read along N, transpose + fp32->bf16 into LDS
      const int n = nb + (tid & 127), kh = tid >> 7;
#pragma unroll
      for (int it = 0; it < 16; ++it) {
        const int kk = it * 2 + kh, k = k0 + kk;
        const float v = (k < DFF) ? W_down[((size_t)e * DFF + k) * HDIM + n] : 0.f;
        Bs[(tid & 127) * KT + kk] = (bf16)v;
      }
      if (k0 + KT + 1 < DFF)
        __builtin_prefetch(W_down + ((size_t)e * DFF + k0 + KT + kh) * HDIM + n, 0, 1);
    }
    asm volatile("s_wait_asynccnt 0x0" ::: "memory");  // own async copies done
    __syncthreads();                                    // publish across waves
    v16bf af[4], bf2[2];
#pragma unroll
    for (int f = 0; f < 4; ++f) af[f] = load_frag16(As, wm * 64 + f * 16, lane);
#pragma unroll
    for (int j = 0; j < 2; ++j) bf2[j] = load_frag16(Bs, wn * 32 + j * 16, lane);
#pragma unroll
    for (int f = 0; f < 4; ++f)
#pragma unroll
      for (int j = 0; j < 2; ++j)
        acc[f][j] = __builtin_amdgcn_wmma_f32_16x16x32_bf16(
            false, af[f], false, bf2[j], (short)0, acc[f][j], false, false);
    __syncthreads();
  }
#pragma unroll
  for (int j = 0; j < 2; ++j) {
    const int n = nb + wn * 32 + j * 16 + (lane & 15);
    const float bias = b_down[e * HDIM + n];
#pragma unroll
    for (int f = 0; f < 4; ++f) {
      const int ml0 = wm * 64 + f * 16 + (lane >> 4) * 8;  // local row in tile
#pragma unroll
      for (int rr = 0; rr < 8; ++rr) {
        const int ml = ml0 + rr;
        const float outv = acc[f][j][rr] + bias;
        atomicAdd(&y[(size_t)tok[ml] * HDIM + n], gv[ml] * outv);
      }
    }
  }
}

// --------------------------------------------------------------------- launch
extern "C" void kernel_launch(void* const* d_in, const int* in_sizes, int n_in,
                              void* d_out, int out_size, void* d_ws, size_t ws_size,
                              hipStream_t stream) {
  (void)in_sizes; (void)n_in; (void)out_size; (void)ws_size;
  const float* x   = (const float*)d_in[0];
  const float* Wr  = (const float*)d_in[1];
  const float* br  = (const float*)d_in[2];
  const float* Wup = (const float*)d_in[3];
  const float* bup = (const float*)d_in[4];
  const float* Wdn = (const float*)d_in[5];
  const float* bdn = (const float*)d_in[6];
  float* y = (float*)d_out;

  // workspace layout (all 256B-aligned): gatesT 4MiB | idx 128KiB | gval 128KiB | act ~88MiB
  char* ws = (char*)d_ws;
  float* gatesT = (float*)ws;
  int*   idx    = (int*)  (ws + (size_t)NEXP * BS * 4);
  float* gval   = (float*)(ws + (size_t)NEXP * BS * 4 + (size_t)NEXP * KSEL * 4);
  bf16*  act    = (bf16*) (ws + (size_t)NEXP * BS * 4 + (size_t)2 * NEXP * KSEL * 4);

  zero_f4<<<(BS * HDIM / 4) / 256, 256, 0, stream>>>((float4*)y);
  router_softmax<<<BS / 128, 256, 0, stream>>>(x, Wr, br, gatesT);
  topk_select<<<NEXP, 1024, 0, stream>>>(gatesT, idx, gval);
  up_silu<<<dim3(DFFP / 128, KSEL / 128, NEXP), 256, 0, stream>>>(x, Wup, bup, idx, act);
  down_combine<<<dim3(HDIM / 128, KSEL / 128, NEXP), 256, 0, stream>>>(act, Wdn, bdn, idx, gval, y);
}
